// QuantAlternatingAttention_79182017069508
// MI455X (gfx1250) — 8-bit matrix operand order repaired and checked by reading, compiled, not run on hardware
//
#include <hip/hip_runtime.h>
#include <hip/hip_fp16.h>

typedef __attribute__((ext_vector_type(16))) _Float16 v16h;
typedef __attribute__((ext_vector_type(8)))  float    v8f;
typedef __attribute__((ext_vector_type(8)))  int      v8i;

#define LBATCH 128
#define TLEN   256
#define NHEAD  8
#define HDIM   64
#define DMODEL 512
#define E3     1536
#define MROWS  32768
#define NQKV   (MROWS * E3)
#define NOUT   (MROWS * DMODEL)

__device__ __constant__ int kLUTtail[17] = {
    1, 5, 13, 37, 103, 280, 761, 2070, 5628, 15298, 41586,
    113043, 307285, 835288, 2270549, 6171992, 16777216};

__device__ __forceinline__ int lut_lookup(int idx) {
  return (idx < 112) ? 0 : kLUTtail[idx - 112];
}

__device__ __forceinline__ void atomicMaxF(float* addr, float v) {
  atomicMax((unsigned int*)addr, __float_as_uint(v));
}

__device__ __forceinline__ void async_copy_b128(const void* gptr, void* lptr) {
  unsigned lds = (unsigned)(unsigned long long)lptr;
  unsigned long long ga = (unsigned long long)gptr;
  asm volatile("global_load_async_to_lds_b128 %0, %1, off"
               :: "v"(lds), "v"(ga) : "memory");
}
__device__ __forceinline__ void wait_async0() {
  asm volatile("s_wait_asynccnt 0x0" ::: "memory");
}

__device__ __forceinline__ int KA8(int v, int hi) {
  return ((v >= 4) ? 32 : 0) + ((v & 2) ? 16 : 0) + ((v & 1) ? 4 : 0) + (hi ? 8 : 0);
}
__device__ __forceinline__ int KB8(int v, int hi) {
  return ((v >= 4) ? 32 : 0) + ((v & 2) ? 16 : 0) + ((v & 1) ? 4 : 0) + (hi ? 8 : 0);
}
__device__ __forceinline__ int KH16(int v, int hi) {
  return ((v >= 4) ? 16 : 0) + 2 * (v & 3) + (hi ? 8 : 0);
}

union FragH { v16h v; unsigned u[8]; };
union H8 { _Float16 h[8]; uint4 q; };

__global__ void init_scalars_kernel(float* scal) {
  if (threadIdx.x < 8) scal[threadIdx.x] = 0.0f;
}

__global__ void absmax_kernel(const float* __restrict__ src, int n, float* slot) {
  int stride = gridDim.x * blockDim.x;
  float m = 0.0f;
  for (int i = blockIdx.x * blockDim.x + threadIdx.x; i < n; i += stride)
    m = fmaxf(m, fabsf(src[i]));
  atomicMaxF(slot, m);
}

__global__ void quantw_kernel(const float* __restrict__ w, _Float16* __restrict__ wq,
                              int n, const float* __restrict__ slot) {
  float s = fmaxf(slot[0] / 127.0f, 1e-8f);
  int stride = gridDim.x * blockDim.x;
  for (int i = blockIdx.x * blockDim.x + threadIdx.x; i < n; i += stride) {
    float q = rintf(w[i] / s);
    q = fminf(fmaxf(q, -128.0f), 127.0f);
    wq[i] = (_Float16)(q * s);
  }
}

template <int MODE>
__global__ __launch_bounds__(256)
void gemm_f16_kernel(const float* __restrict__ Afp32,
                     const _Float16* __restrict__ Af16,
                     const _Float16* __restrict__ Bw,
                     const float* __restrict__ bias,
                     float* __restrict__ Cout,
                     int N, float* absSlot) {
  __shared__ _Float16 As[64][40];
  __shared__ _Float16 Bs[64][40];

  const int m0 = blockIdx.x * 64;
  const int n0 = blockIdx.y * 64;
  const int tid = threadIdx.x;
  const int wave = tid >> 5;
  const int lane = tid & 31;
  const int hi = (lane >= 16);
  const int n16 = lane & 15;
  const int mt = wave & 3;

  const int sr = tid >> 2;
  const int sseg = (tid & 3) * 8;

  v8f acc[2] = {};

  for (int kk = 0; kk < 512; kk += 32) {
    async_copy_b128(&Bw[(size_t)(n0 + sr) * 512 + kk + sseg], &Bs[sr][sseg]);

    if (MODE == 0) {
      const float4* ap =
          reinterpret_cast<const float4*>(&Afp32[(size_t)(m0 + sr) * 512 + kk + sseg]);
      float4 a0 = ap[0];
      float4 a1 = ap[1];
      H8 pk;
      pk.h[0] = (_Float16)a0.x; pk.h[1] = (_Float16)a0.y;
      pk.h[2] = (_Float16)a0.z; pk.h[3] = (_Float16)a0.w;
      pk.h[4] = (_Float16)a1.x; pk.h[5] = (_Float16)a1.y;
      pk.h[6] = (_Float16)a1.z; pk.h[7] = (_Float16)a1.w;
      *reinterpret_cast<uint4*>(&As[sr][sseg]) = pk.q;
    } else {
      async_copy_b128(&Af16[(size_t)(m0 + sr) * 512 + kk + sseg], &As[sr][sseg]);
    }

    if (kk + 32 < 512) {
      __builtin_prefetch(&Bw[(size_t)(n0 + sr) * 512 + kk + 32 + sseg], 0, 1);
      if (MODE == 0)
        __builtin_prefetch(&Afp32[(size_t)(m0 + sr) * 512 + kk + 32 + sseg], 0, 1);
      else
        __builtin_prefetch(&Af16[(size_t)(m0 + sr) * 512 + kk + 32 + sseg], 0, 1);
    }

    wait_async0();
    __syncthreads();

    FragH fa;
    for (int v = 0; v < 8; ++v)
      fa.u[v] = *(const unsigned*)&As[mt * 16 + n16][KH16(v, hi)];

    for (int j = 0; j < 2; ++j) {
      int nt = 2 * (wave >> 2) + j;
      FragH fb;
      for (int v = 0; v < 8; ++v)
        fb.u[v] = *(const unsigned*)&Bs[nt * 16 + n16][KH16(v, hi)];
      acc[j] = __builtin_amdgcn_wmma_f32_16x16x32_f16(
          false, fa.v, false, fb.v, (short)0, acc[j], false, false);
    }
    __syncthreads();
  }

  float amax = 0.0f;
  for (int j = 0; j < 2; ++j) {
    int nt = 2 * (wave >> 2) + j;
    int n_g = n0 + nt * 16 + n16;
    float b = bias[n_g];
    for (int r = 0; r < 8; ++r) {
      int m_g = m0 + mt * 16 + r + (hi ? 8 : 0);
      float val = acc[j][r] + b;
      Cout[(size_t)m_g * N + n_g] = val;
      amax = fmaxf(amax, fabsf(val));
    }
  }
  if (MODE == 1) atomicMaxF(absSlot, amax);
}

__global__ void qkv_absmax_kernel(const float* __restrict__ qkv, float* scal) {
  int stride = gridDim.x * blockDim.x;
  float mq = 0.0f, mk = 0.0f, mv = 0.0f;
  for (int i = blockIdx.x * blockDim.x + threadIdx.x; i < NQKV; i += stride) {
    int col = i - (i / E3) * E3;
    float a = fabsf(qkv[i]);
    if (col < 512)       mq = fmaxf(mq, a * 0.125f);
    else if (col < 1024) mk = fmaxf(mk, a);
    else                 mv = fmaxf(mv, a);
  }
  atomicMaxF(&scal[2], mq);
  atomicMaxF(&scal[3], mk);
  atomicMaxF(&scal[4], mv);
}

__global__ __launch_bounds__(128)
void attention_kernel(const float* __restrict__ qkv,
                      _Float16* __restrict__ ctx,
                      const float* __restrict__ scal) {
  __shared__ signed char k8[256][64];
  __shared__ signed char q8[64][64];
  __shared__ _Float16 vlds[64][264];
  __shared__ _Float16 pstage[4][16][40];

  const int bid = blockIdx.x;
  const int qb = bid & 3;
  const int h = (bid >> 2) & 7;
  const int lb = bid >> 5;
  const int qbase = qb * 64;

  const float sq = fmaxf(scal[2] / 127.0f, 1e-8f);
  const float sk = fmaxf(scal[3] / 127.0f, 1e-8f);
  const float sv = fmaxf(scal[4] / 127.0f, 1e-8f);
  const float score_scale = sq * sk;

  const int tid = threadIdx.x;
  for (int i = tid; i < 256 * 16; i += 128) {
    int t = i >> 4, d4 = (i & 15) * 4;
    size_t base = (size_t)(lb * 256 + t) * E3 + h * 64 + d4;
    float4 kv = *reinterpret_cast<const float4*>(&qkv[base + 512]);
    float4 vv = *reinterpret_cast<const float4*>(&qkv[base + 1024]);
    float kq[4] = {kv.x, kv.y, kv.z, kv.w};
    float vq[4] = {vv.x, vv.y, vv.z, vv.w};
    int packed = 0;
    for (int j = 0; j < 4; ++j) {
      int ki = (int)fminf(fmaxf(rintf(kq[j] / sk), -128.0f), 127.0f);
      packed |= (ki & 0xff) << (8 * j);
      float vi = fminf(fmaxf(rintf(vq[j] / sv), -128.0f), 127.0f);
      vlds[d4 + j][t] = (_Float16)(vi * sv);
    }
    *reinterpret_cast<int*>(&k8[t][d4]) = packed;
  }
  for (int i = tid; i < 64 * 16; i += 128) {
    int tl = i >> 4, d4 = (i & 15) * 4;
    size_t base = (size_t)(lb * 256 + qbase + tl) * E3 + h * 64 + d4;
    float4 qv = *reinterpret_cast<const float4*>(&qkv[base]);
    float qq[4] = {qv.x, qv.y, qv.z, qv.w};
    int packed = 0;
    for (int j = 0; j < 4; ++j) {
      int qi = (int)fminf(fmaxf(rintf(qq[j] * 0.125f / sq), -128.0f), 127.0f);
      packed |= (qi & 0xff) << (8 * j);
    }
    *reinterpret_cast<int*>(&q8[tl][d4]) = packed;
  }
  __syncthreads();

  const int wave = tid >> 5;
  const int lane = tid & 31;
  const int hi = (lane >= 16);
  const int n16 = lane & 15;

  v8i aq;
  for (int v = 0; v < 8; ++v)
    aq[v] = *(const int*)&q8[wave * 16 + n16][KA8(v, hi)];

  int rm[8];
  for (int r = 0; r < 8; ++r) rm[r] = (int)0x80000000;
  for (int kt = 0; kt < 16; ++kt) {
    v8i bk;
    for (int v = 0; v < 8; ++v)
      bk[v] = *(const int*)&k8[kt * 16 + n16][KB8(v, hi)];
    v8i cz = {};
    v8i c = __builtin_amdgcn_wmma_i32_16x16x64_iu8(true, aq, true, bk, cz, false, false);
    for (int r = 0; r < 8; ++r) rm[r] = max(rm[r], c[r]);
  }
  for (int r = 0; r < 8; ++r)
    for (int m = 1; m <= 8; m <<= 1)
      rm[r] = max(rm[r], __shfl_xor(rm[r], m, 32));
  float rowmax[8];
  for (int r = 0; r < 8; ++r) rowmax[r] = score_scale * (float)rm[r];

  float ysum[8] = {0, 0, 0, 0, 0, 0, 0, 0};
  v8f acc[4] = {};
  for (int kt2 = 0; kt2 < 8; ++kt2) {
    for (int sub = 0; sub < 2; ++sub) {
      int kt = kt2 * 2 + sub;
      v8i bk;
      for (int v = 0; v < 8; ++v)
        bk[v] = *(const int*)&k8[kt * 16 + n16][KB8(v, hi)];
      v8i cz = {};
      v8i c = __builtin_amdgcn_wmma_i32_16x16x64_iu8(true, aq, true, bk, cz, false, false);
      for (int r = 0; r < 8; ++r) {
        float s = score_scale * (float)c[r];
        int xi = (int)rintf((s - rowmax[r]) * 16.0f);
        xi = max(xi, -128);
        xi = min(xi, 0);
        float yf = (float)lut_lookup(xi + 128) * (1.0f / 65536.0f);
        ysum[r] += yf;
        pstage[wave][r + (hi ? 8 : 0)][sub * 16 + n16] = (_Float16)yf;
      }
    }
    asm volatile("s_wait_dscnt 0" ::: "memory");

    FragH pa;
    for (int v = 0; v < 8; ++v)
      pa.u[v] = *(const unsigned*)&pstage[wave][n16][KH16(v, hi)];
    for (int nt = 0; nt < 4; ++nt) {
      FragH vb;
      for (int v = 0; v < 8; ++v)
        vb.u[v] = *(const unsigned*)&vlds[nt * 16 + n16][kt2 * 32 + KH16(v, hi)];
      acc[nt] = __builtin_amdgcn_wmma_f32_16x16x32_f16(
          false, pa.v, false, vb.v, (short)0, acc[nt], false, false);
    }
  }

  for (int r = 0; r < 8; ++r)
    for (int m = 1; m <= 8; m <<= 1)
      ysum[r] += __shfl_xor(ysum[r], m, 32);
  float inv[8];
  for (int r = 0; r < 8; ++r)
    inv[r] = 1.0f / fmaxf(ysum[r], 1.0f / 65536.0f);

  for (int nt = 0; nt < 4; ++nt) {
    for (int r = 0; r < 8; ++r) {
      int qrow = qbase + wave * 16 + r + (hi ? 8 : 0);
      int col = h * 64 + nt * 16 + n16;
      float val = acc[nt][r] * inv[r];
      ctx[(size_t)(lb * 256 + qrow) * 512 + col] = (_Float16)val;
    }
  }
}

__global__ void quant_out_kernel(const float* __restrict__ src, float* __restrict__ dst,
                                 const float* __restrict__ slot) {
  float s = fmaxf(slot[0] / 127.0f, 1e-8f);
  int stride = gridDim.x * blockDim.x;
  for (int i = blockIdx.x * blockDim.x + threadIdx.x; i < NOUT; i += stride) {
    float q = rintf(src[i] / s);
    q = fminf(fmaxf(q, -128.0f), 127.0f);
    dst[i] = q * s;
  }
}

extern "C" void kernel_launch(void* const* d_in, const int* in_sizes, int n_in,
                              void* d_out, int out_size, void* d_ws, size_t ws_size,
                              hipStream_t stream) {
  const float* x     = (const float*)d_in[0];
  const float* W_qkv = (const float*)d_in[1];
  const float* b_qkv = (const float*)d_in[2];
  const float* W_out = (const float*)d_in[3];
  const float* b_out = (const float*)d_in[4];

  char* ws = (char*)d_ws;
  const size_t OFF_SCAL = 0;
  const size_t OFF_WQ   = 256;
  const size_t OFF_WO   = OFF_WQ + (size_t)E3 * DMODEL * 2;
  const size_t OFF_QKV  = OFF_WO + (size_t)DMODEL * DMODEL * 2;
  const size_t OFF_CTX  = OFF_QKV + (size_t)NQKV * 4;
  const size_t OFF_OUT  = OFF_QKV;

  float*     scal   = (float*)(ws + OFF_SCAL);
  _Float16*  Wq16   = (_Float16*)(ws + OFF_WQ);
  _Float16*  Wo16   = (_Float16*)(ws + OFF_WO);
  float*     qkvbuf = (float*)(ws + OFF_QKV);
  _Float16*  ctx16  = (_Float16*)(ws + OFF_CTX);
  float*     outbuf = (float*)(ws + OFF_OUT);

  init_scalars_kernel<<<1, 32, 0, stream>>>(scal);
  absmax_kernel<<<512, 256, 0, stream>>>(W_qkv, E3 * DMODEL, &scal[0]);
  absmax_kernel<<<256, 256, 0, stream>>>(W_out, DMODEL * DMODEL, &scal[1]);
  quantw_kernel<<<512, 256, 0, stream>>>(W_qkv, Wq16, E3 * DMODEL, &scal[0]);
  quantw_kernel<<<256, 256, 0, stream>>>(W_out, Wo16, DMODEL * DMODEL, &scal[1]);

  gemm_f16_kernel<0><<<dim3(MROWS / 64, E3 / 64), 256, 0, stream>>>(
      x, nullptr, Wq16, b_qkv, qkvbuf, E3, nullptr);

  qkv_absmax_kernel<<<2048, 256, 0, stream>>>(qkvbuf, scal);

  attention_kernel<<<LBATCH * NHEAD * (TLEN / 64), 128, 0, stream>>>(qkvbuf, ctx16, scal);

  gemm_f16_kernel<1><<<dim3(MROWS / 64, DMODEL / 64), 256, 0, stream>>>(
      nullptr, ctx16, Wo16, b_out, outbuf, DMODEL, &scal[5]);

  quant_out_kernel<<<4096, 256, 0, stream>>>(outbuf, (float*)d_out, &scal[5]);
}
